// GAT_50577534877738
// MI455X (gfx1250) — matrix operand order repaired and checked by reading, not compiled, not run on hardware
//
#include <hip/hip_runtime.h>

// ---------------------------------------------------------------------------
// GAT forward for MI455X (gfx1250), wave32 + v_wmma_f32_16x16x32_bf16.
// N=4096 nodes, IN=256, HID=256, HEADS=4, CLS=64, dense masked softmax.
// All heavy matmuls (x@W1, attn@g per head, h@W2, attn2@g2) run on WMMA
// with bf16 operands / f32 accumulation. Probability tiles are built
// directly in A-fragment register layout (vectorized adj/t loads + exp);
// g is pre-transposed so B-fragments are two contiguous b128 loads per lane.
// GEMM: 16x64 tile per wave (A-fragment reuse). Attention L1: 16x128/wave.
// ---------------------------------------------------------------------------

#define NNODES 4096
#define INF_   256
#define HID_   256
#define HEADS_ 4
#define CLS_   64
#define SLOPE_ 0.2f

typedef unsigned short u16;
typedef __attribute__((ext_vector_type(16))) __bf16 bf16x16;
typedef __attribute__((ext_vector_type(8)))  float  f32x8;

union FragAB { uint4 q[2]; bf16x16 v; u16 us[16]; };
union FragC  { f32x8 v; float f[8]; };

__device__ __forceinline__ u16 f2bf(float f) {            // RNE f32 -> bf16
  unsigned u = __float_as_uint(f);
  u += 0x7FFFu + ((u >> 16) & 1u);
  return (u16)(u >> 16);
}
__device__ __forceinline__ float bf2f(u16 b) {
  return __uint_as_float(((unsigned)b) << 16);
}

// -------------------------- elementwise convert ----------------------------
__global__ void k_f2b(const float* __restrict__ src, u16* __restrict__ dst, int n) {
  int i = blockIdx.x * blockDim.x + threadIdx.x;
  if (i < n) dst[i] = f2bf(src[i]);
}

// -------------------- tiled transpose (f32 -> bf16) ------------------------
__global__ void k_tr_f2b(const float* __restrict__ src, u16* __restrict__ dst,
                         int R, int C) {  // src[R][C] -> dst[C][R]
  __shared__ float tile[32][33];
  int c0 = blockIdx.x * 32, r0 = blockIdx.y * 32;
  int tx = threadIdx.x, ty = threadIdx.y;
  for (int yy = ty; yy < 32; yy += 8) {
    int r = r0 + yy, c = c0 + tx;
    tile[yy][tx] = (r < R && c < C) ? src[(size_t)r * C + c] : 0.f;
  }
  __syncthreads();
  for (int yy = ty; yy < 32; yy += 8) {
    int c = c0 + yy, r = r0 + tx;
    if (c < C && r < R) dst[(size_t)c * R + r] = f2bf(tile[tx][yy]);
  }
}

// -------------------- tiled transpose (bf16 -> bf16) -----------------------
__global__ void k_tr_b(const u16* __restrict__ src, u16* __restrict__ dst,
                       int R, int C) {  // src[R][C] -> dst[C][R]
  __shared__ u16 tile[32][34];
  int c0 = blockIdx.x * 32, r0 = blockIdx.y * 32;
  int tx = threadIdx.x, ty = threadIdx.y;
  for (int yy = ty; yy < 32; yy += 8) {
    int r = r0 + yy, c = c0 + tx;
    tile[yy][tx] = (r < R && c < C) ? src[(size_t)r * C + c] : (u16)0;
  }
  __syncthreads();
  for (int yy = ty; yy < 32; yy += 8) {
    int c = c0 + yy, r = r0 + tx;
    if (c < C && r < R) dst[(size_t)c * R + r] = tile[tx][yy];
  }
}

// ------------------------------ WMMA GEMM ----------------------------------
// C_bf16[M][Nn] = A_bf16[M][K] @ Bt_bf16[Nn][K]^T.
// One 16x64 tile per wave: A-fragment reused across 4 WMMAs per K-step.
__global__ void k_gemm_bf16(const u16* __restrict__ A, const u16* __restrict__ Bt,
                            u16* __restrict__ C, int M, int Nn, int K) {
  int wid  = blockIdx.x * (blockDim.x >> 5) + (threadIdx.x >> 5);
  int lane = threadIdx.x & 31;
  int n64 = Nn >> 6;
  int mt = wid / n64, nc = wid % n64;
  if (mt >= (M >> 4)) return;                 // wave-uniform
  int r = lane & 15, kh = lane >> 4;
  const u16* arow = A + (size_t)(mt * 16 + r) * K;
  FragC acc[4];
#pragma unroll
  for (int ff = 0; ff < 4; ++ff)
#pragma unroll
    for (int v = 0; v < 8; ++v) acc[ff].f[v] = 0.f;

  for (int k0 = 0; k0 < K; k0 += 32) {
    FragAB a;
    a.q[0] = *reinterpret_cast<const uint4*>(arow + k0 + kh * 8);
    a.q[1] = *reinterpret_cast<const uint4*>(arow + k0 + 16 + kh * 8);
    if (k0 + 32 < K) __builtin_prefetch(arow + k0 + 32, 0, 1);
#pragma unroll
    for (int ff = 0; ff < 4; ++ff) {
      FragAB b;
      const u16* br = Bt + (size_t)(nc * 64 + ff * 16 + r) * K + k0 + kh * 8;
      b.q[0] = *reinterpret_cast<const uint4*>(br);
      b.q[1] = *reinterpret_cast<const uint4*>(br + 16);
      acc[ff].v = __builtin_amdgcn_wmma_f32_16x16x32_bf16(false, a.v, false, b.v,
                                                          (short)0, acc[ff].v,
                                                          false, false);
    }
  }
#pragma unroll
  for (int ff = 0; ff < 4; ++ff) {
    int col = nc * 64 + ff * 16 + r;
#pragma unroll
    for (int v = 0; v < 8; ++v) {
      int row = mt * 16 + kh * 8 + v;
      C[(size_t)row * Nn + col] = f2bf(acc[ff].f[v]);
    }
  }
}

// ------------------- per-(node,head) src/dst score dots --------------------
__global__ void k_src_dst(const u16* __restrict__ g, const float* __restrict__ aS,
                          const float* __restrict__ aD, float* __restrict__ s,
                          float* __restrict__ t, int Nn, int H, int D) {
  int idx = blockIdx.x * blockDim.x + threadIdx.x;
  if (idx >= Nn * H) return;
  int n = idx % Nn, h = idx / Nn;
  const u16* gr = g + (size_t)n * (H * D) + (size_t)h * D;
  float ss = 0.f, tt = 0.f;
  for (int d = 0; d < D; d += 8) {           // 8 bf16 per b128 load
    uint4 q = *reinterpret_cast<const uint4*>(gr + d);
    unsigned w[4] = {q.x, q.y, q.z, q.w};
#pragma unroll
    for (int p = 0; p < 4; ++p) {
      float v0 = bf2f((u16)(w[p] & 0xFFFFu));
      float v1 = bf2f((u16)(w[p] >> 16));
      ss += v0 * aS[d + 2 * p] + v1 * aS[d + 2 * p + 1];
      tt += v0 * aD[d + 2 * p] + v1 * aD[d + 2 * p + 1];
    }
  }
  s[(size_t)h * Nn + n] = ss;
  t[(size_t)h * Nn + n] = tt;
}

// -------------- masked online-softmax row stats (max, 1/sum) ---------------
__global__ void k_row_stats(const float* __restrict__ sarr, const float* __restrict__ tarr,
                            const unsigned char* __restrict__ adj,
                            float* __restrict__ stats, int Nn, int H) {
  int wid  = blockIdx.x * (blockDim.x >> 5) + (threadIdx.x >> 5);
  int lane = threadIdx.x & 31;
  int i = wid % Nn, h = wid / Nn;
  if (h >= H) return;                          // wave-uniform
  float si = sarr[(size_t)h * Nn + i];
  const float* tp = tarr + (size_t)h * Nn;
  const unsigned char* ar = adj + (size_t)i * Nn;
  float m = -1e30f, sum = 0.f;
  for (int j = lane; j < Nn; j += 32) {
    if (!ar[j]) continue;
    float e = si + tp[j];
    e = (e >= 0.f) ? e : SLOPE_ * e;
    float mn = fmaxf(m, e);
    sum = sum * __expf(m - mn) + __expf(e - mn);
    m = mn;
  }
  for (int off = 16; off >= 1; off >>= 1) {
    float m2 = __shfl_xor(m, off, 32);
    float s2 = __shfl_xor(sum, off, 32);
    float mn = fmaxf(m, m2);
    sum = sum * __expf(m - mn) + s2 * __expf(m2 - mn);
    m = mn;
  }
  if (lane == 0) {
    stats[((size_t)h * Nn + i) * 2 + 0] = m;
    stats[((size_t)h * Nn + i) * 2 + 1] = 1.f / sum;
  }
}

// ---------------------- fused attention aggregation ------------------------
// Per wave: one head, 16 output rows x (NF*16) output cols. Probability tile
// is built directly in A-fragment layout from vectorized adjacency/t loads;
// B-fragments come from gt (feature-major transpose of g), contiguous loads.
__device__ __forceinline__ u16 pbf(float tv, unsigned mbyte,
                                   float sI, float mI, float invI) {
  float e = sI + tv;
  e = (e >= 0.f) ? e : SLOPE_ * e;
  float p = (mbyte & 0xFFu) ? __expf(e - mI) * invI : 0.f;  // exp(-1e9-m)==0
  return f2bf(p);
}

template <int NF>
__global__ void k_attn_wmma(const float* __restrict__ sarr, const float* __restrict__ tarr,
                            const float* __restrict__ stats,
                            const unsigned char* __restrict__ adj,
                            const u16* __restrict__ gt, float* __restrict__ out,
                            int Nn, int D) {
  int lane = threadIdx.x;
  int r = lane & 15, kh = lane >> 4;
  int i0 = blockIdx.x * 16;
  int f0 = blockIdx.y * (NF * 16);
  int h  = blockIdx.z;
  int i  = i0 + r;
  float sI   = sarr[(size_t)h * Nn + i];
  float mI   = stats[((size_t)h * Nn + i) * 2 + 0];
  float invI = stats[((size_t)h * Nn + i) * 2 + 1];
  const float* tp = tarr + (size_t)h * Nn;
  const unsigned char* ar = adj + (size_t)i * Nn;

  FragC acc[NF];
#pragma unroll
  for (int ff = 0; ff < NF; ++ff)
#pragma unroll
    for (int v = 0; v < 8; ++v) acc[ff].f[v] = 0.f;

  for (int j0 = 0; j0 < Nn; j0 += 32) {
    int jb0 = j0 + kh * 8;        // K = jb0..jb0+7     (A-frag VGPR0..3)
    int jb1 = j0 + 16 + kh * 8;   // K = jb1..jb1+7     (A-frag VGPR4..7)
    uint2  m0  = *reinterpret_cast<const uint2*>(ar + jb0);   // 8 adj bytes
    uint2  m1  = *reinterpret_cast<const uint2*>(ar + jb1);
    float4 t0a = *reinterpret_cast<const float4*>(tp + jb0);
    float4 t0b = *reinterpret_cast<const float4*>(tp + jb0 + 4);
    float4 t1a = *reinterpret_cast<const float4*>(tp + jb1);
    float4 t1b = *reinterpret_cast<const float4*>(tp + jb1 + 4);
    FragAB A;
    A.us[0]  = pbf(t0a.x, m0.x,       sI, mI, invI);
    A.us[1]  = pbf(t0a.y, m0.x >> 8,  sI, mI, invI);
    A.us[2]  = pbf(t0a.z, m0.x >> 16, sI, mI, invI);
    A.us[3]  = pbf(t0a.w, m0.x >> 24, sI, mI, invI);
    A.us[4]  = pbf(t0b.x, m0.y,       sI, mI, invI);
    A.us[5]  = pbf(t0b.y, m0.y >> 8,  sI, mI, invI);
    A.us[6]  = pbf(t0b.z, m0.y >> 16, sI, mI, invI);
    A.us[7]  = pbf(t0b.w, m0.y >> 24, sI, mI, invI);
    A.us[8]  = pbf(t1a.x, m1.x,       sI, mI, invI);
    A.us[9]  = pbf(t1a.y, m1.x >> 8,  sI, mI, invI);
    A.us[10] = pbf(t1a.z, m1.x >> 16, sI, mI, invI);
    A.us[11] = pbf(t1a.w, m1.x >> 24, sI, mI, invI);
    A.us[12] = pbf(t1b.x, m1.y,       sI, mI, invI);
    A.us[13] = pbf(t1b.y, m1.y >> 8,  sI, mI, invI);
    A.us[14] = pbf(t1b.z, m1.y >> 16, sI, mI, invI);
    A.us[15] = pbf(t1b.w, m1.y >> 24, sI, mI, invI);
#pragma unroll
    for (int ff = 0; ff < NF; ++ff) {
      FragAB B;
      const u16* grow = gt + (size_t)(h * D + f0 + ff * 16 + r) * Nn + j0 + kh * 8;
      B.q[0] = *reinterpret_cast<const uint4*>(grow);
      B.q[1] = *reinterpret_cast<const uint4*>(grow + 16);
      acc[ff].v = __builtin_amdgcn_wmma_f32_16x16x32_bf16(false, A.v, false, B.v,
                                                          (short)0, acc[ff].v,
                                                          false, false);
    }
  }
#pragma unroll
  for (int ff = 0; ff < NF; ++ff) {
    int col = f0 + ff * 16 + r;
#pragma unroll
    for (int v = 0; v < 8; ++v) {
      int row = i0 + kh * 8 + v;
      out[((size_t)h * Nn + row) * D + col] = acc[ff].f[v];
    }
  }
}

// --------------------------- head mean + ELU -------------------------------
__global__ void k_mean_elu(const float* __restrict__ att, u16* __restrict__ hb, int Nn) {
  int idx = blockIdx.x * blockDim.x + threadIdx.x;
  if (idx >= Nn * HID_) return;
  float v = 0.f;
#pragma unroll
  for (int h = 0; h < HEADS_; ++h) v += att[(size_t)h * Nn * HID_ + idx];
  v *= 0.25f;
  v = (v > 0.f) ? v : (__expf(v) - 1.f);       // ELU
  hb[idx] = f2bf(v);
}

// ---------------------------------------------------------------------------
extern "C" void kernel_launch(void* const* d_in, const int* in_sizes, int n_in,
                              void* d_out, int out_size, void* d_ws, size_t ws_size,
                              hipStream_t stream) {
  (void)in_sizes; (void)n_in; (void)out_size; (void)ws_size;
  const float*         x   = (const float*)d_in[0];
  const unsigned char* adj = (const unsigned char*)d_in[1];   // bool mask
  const float*         W1  = (const float*)d_in[2];
  const float*         a1s = (const float*)d_in[3];
  const float*         a1d = (const float*)d_in[4];
  const float*         W2  = (const float*)d_in[5];
  const float*         a2s = (const float*)d_in[6];
  const float*         a2d = (const float*)d_in[7];
  float* out = (float*)d_out;

  char* base = (char*)d_ws;
  size_t off = 0;
  auto alloc = [&](size_t bytes) -> void* {
    void* r = base + off;
    off = (off + bytes + 255) & ~(size_t)255;
    return r;
  };
  u16*   xb    = (u16*)  alloc((size_t)NNODES * INF_ * 2);            // x bf16
  u16*   w1t   = (u16*)  alloc((size_t)HEADS_ * HID_ * INF_ * 2);     // W1^T bf16 [1024][256]
  u16*   g1b   = (u16*)  alloc((size_t)NNODES * HEADS_ * HID_ * 2);   // g1 bf16 [4096][1024]
  u16*   g1t   = (u16*)  alloc((size_t)HEADS_ * HID_ * NNODES * 2);   // g1^T bf16 [1024][4096]
  float* s1    = (float*)alloc((size_t)HEADS_ * NNODES * 4);
  float* t1    = (float*)alloc((size_t)HEADS_ * NNODES * 4);
  float* st1   = (float*)alloc((size_t)HEADS_ * NNODES * 2 * 4);
  float* att1  = (float*)alloc((size_t)HEADS_ * NNODES * HID_ * 4);   // per-head layer1 out
  u16*   hb    = (u16*)  alloc((size_t)NNODES * HID_ * 2);            // ELU(mean) bf16
  u16*   w2t   = (u16*)  alloc((size_t)CLS_ * HID_ * 2);              // W2^T bf16 [64][256]
  u16*   g2b   = (u16*)  alloc((size_t)NNODES * CLS_ * 2);
  u16*   g2t   = (u16*)  alloc((size_t)CLS_ * NNODES * 2);
  float* s2    = (float*)alloc((size_t)NNODES * 4);
  float* t2    = (float*)alloc((size_t)NNODES * 4);
  float* st2   = (float*)alloc((size_t)NNODES * 2 * 4);

  // ---- layer 1 ----
  k_f2b<<<(NNODES * INF_ + 255) / 256, 256, 0, stream>>>(x, xb, NNODES * INF_);
  k_tr_f2b<<<dim3((HEADS_ * HID_) / 32, INF_ / 32), dim3(32, 8), 0, stream>>>(
      W1, w1t, INF_, HEADS_ * HID_);
  {   // g1 = x @ W1 : tiles = 256 * (1024/64) = 4096 waves, 4 waves/block
    int tiles = (NNODES / 16) * ((HEADS_ * HID_) / 64);
    k_gemm_bf16<<<tiles / 4, 128, 0, stream>>>(xb, w1t, g1b, NNODES, HEADS_ * HID_, INF_);
  }
  k_tr_b<<<dim3((HEADS_ * HID_) / 32, NNODES / 32), dim3(32, 8), 0, stream>>>(
      g1b, g1t, NNODES, HEADS_ * HID_);
  k_src_dst<<<(NNODES * HEADS_ + 255) / 256, 256, 0, stream>>>(
      g1b, a1s, a1d, s1, t1, NNODES, HEADS_, HID_);
  k_row_stats<<<(NNODES * HEADS_) / 8, 256, 0, stream>>>(s1, t1, adj, st1, NNODES, HEADS_);
  k_attn_wmma<8><<<dim3(NNODES / 16, HID_ / 128, HEADS_), 32, 0, stream>>>(
      s1, t1, st1, adj, g1t, att1, NNODES, HID_);
  k_mean_elu<<<(NNODES * HID_ + 255) / 256, 256, 0, stream>>>(att1, hb, NNODES);

  // ---- layer 2 ----
  k_tr_f2b<<<dim3(CLS_ / 32, HID_ / 32), dim3(32, 8), 0, stream>>>(W2, w2t, HID_, CLS_);
  {   // g2 = h @ W2 : tiles = 256 * (64/64) = 256 waves
    int tiles = (NNODES / 16) * (CLS_ / 64);
    k_gemm_bf16<<<tiles / 4, 128, 0, stream>>>(hb, w2t, g2b, NNODES, CLS_, HID_);
  }
  k_tr_b<<<dim3(CLS_ / 32, NNODES / 32), dim3(32, 8), 0, stream>>>(g2b, g2t, NNODES, CLS_);
  k_src_dst<<<(NNODES + 255) / 256, 256, 0, stream>>>(g2b, a2s, a2d, s2, t2, NNODES, 1, CLS_);
  k_row_stats<<<NNODES / 8, 256, 0, stream>>>(s2, t2, adj, st2, NNODES, 1);
  k_attn_wmma<4><<<dim3(NNODES / 16, CLS_ / 64, 1), 32, 0, stream>>>(
      s2, t2, st2, adj, g2t, out, NNODES, CLS_);
}
